// VQEmbedding_3874060501677
// MI455X (gfx1250) — matrix operand order repaired and checked by reading, not compiled, not run on hardware
//
#include <hip/hip_runtime.h>

typedef __attribute__((ext_vector_type(16))) __bf16          v16bf;
typedef __attribute__((ext_vector_type(8)))  float           v8f;
typedef __attribute__((ext_vector_type(8)))  unsigned short  us8;
typedef __attribute__((ext_vector_type(4)))  unsigned short  us4;
typedef __attribute__((ext_vector_type(4)))  unsigned int    u32x4;
typedef __attribute__((ext_vector_type(8)))  unsigned int    u32x8;

union BFV { us8 h[2]; v16bf v; };

__device__ __forceinline__ unsigned short f2bf(float f) {
    unsigned int u = __builtin_bit_cast(unsigned int, f);
    u = (u + 0x7FFFu + ((u >> 16) & 1u)) >> 16;   // round-to-nearest-even
    return (unsigned short)u;
}

#define NROWS_TOTAL 65536      // 32*2048
#define DIM         256
#define NCODES      1024
#define BROWS       64         // rows per block
#define CT          16         // codes per LDS tile (double buffered)
#define NTILES      (NCODES / CT)
#define TILE_ELTS   (CT * DIM) // 4096 bf16 elements = 8KB
#define NQ          16777216   // NROWS_TOTAL*DIM

// Issue one TDM DMA: contiguous TILE_ELTS bf16 elements global -> LDS.
// D# group0/group1 built per CDNA5 ISA 8.3/8.4 (count=1, type=2 "image",
// data_size=1 (2 bytes), 1-D tile: tile_dim0 = tensor_dim0 = TILE_ELTS).
__device__ __forceinline__ void tdm_tile_load(const unsigned short* gsrc,
                                              unsigned short* ldst) {
    unsigned long long ga  = (unsigned long long)(uintptr_t)gsrc;
    unsigned           lds = (unsigned)(uintptr_t)ldst;   // low 32 bits of flat LDS addr = LDS offset
    u32x4 g0;
    g0[0] = 1u;                                        // [1:0] count=1, gather off
    g0[1] = lds;                                       // [63:32]  lds_addr (bytes)
    g0[2] = (unsigned)ga;                              // [95:64]  global_addr low
    g0[3] = (unsigned)(ga >> 32) | 0x80000000u;        // [120:96] addr hi, [127:126] type=2
    u32x8 g1;
    g1[0] = 0x00010000u;                               // workgroup_mask=0, data_size=1 (2B)
    g1[1] = (TILE_ELTS & 0xFFFFu) << 16;               // tensor_dim0[15:0]  @ bits 63:48
    g1[2] = (TILE_ELTS >> 16) | (1u << 16);            // tensor_dim0[31:16]; tensor_dim1=1
    g1[3] = ((unsigned)TILE_ELTS) << 16;               // tile_dim0 @ bits 127:112
    g1[4] = 1u;                                        // tile_dim1=1, tile_dim2=0
    g1[5] = (unsigned)TILE_ELTS;                       // tensor_dim0_stride low 32
    g1[6] = 0u;
    g1[7] = 0u;
    asm volatile("tensor_load_to_lds %0, %1" :: "s"(g0), "s"(g1) : "memory");
}

// ------------------------------------------------------------------ prep ----
// Convert codebook f32 -> bf16 once (reused by all 1024 blocks), plus ||e||^2.
__global__ __launch_bounds__(128) void vq_prep(const float* __restrict__ emb,
                                               unsigned short* __restrict__ ebf,
                                               float* __restrict__ esq) {
    const int base = blockIdx.x * 16 * DIM;            // 16 codes per block
    const float4* eg = (const float4*)(emb + base);
    #pragma unroll
    for (int i = 0; i < 8; ++i) {
        int f = i * 128 + threadIdx.x;                 // 0..1023 float4s
        float4 v = eg[f];
        us4 p = { f2bf(v.x), f2bf(v.y), f2bf(v.z), f2bf(v.w) };
        *(us4*)&ebf[base + f * 4] = p;
    }
    if (threadIdx.x < 16) {
        int m = blockIdx.x * 16 + threadIdx.x;
        const float4* e = (const float4*)(emb + (size_t)m * DIM);
        float s = 0.f;
        #pragma unroll 8
        for (int i = 0; i < DIM / 4; ++i) {
            float4 v = e[i];
            s += v.x * v.x + v.y * v.y + v.z * v.z + v.w * v.w;
        }
        esq[m] = s;
    }
}

// ------------------------------------------------------------------ main ----
__global__ __launch_bounds__(128, 1) void vq_main(const float* __restrict__ x,
                                                  const float* __restrict__ emb,
                                                  const unsigned short* __restrict__ ebf,
                                                  const float* __restrict__ esq,
                                                  float* __restrict__ out,
                                                  float* __restrict__ partial) {
    __shared__ unsigned short xs[BROWS * DIM];     // 32 KB bf16 x tile
    __shared__ unsigned short es[2][TILE_ELTS];    // 2 x 8 KB double-buffered e tiles
    __shared__ int   bidx_s[BROWS];
    __shared__ float lsum_s[4];

    const int tid  = threadIdx.x;
    const int wave = tid >> 5;
    const int lane = tid & 31;
    const int r0   = blockIdx.x * BROWS;

    // ---- stage x rows -> bf16 LDS ----
    {
        const float4* xg = (const float4*)(x + (size_t)r0 * DIM);
        #pragma unroll
        for (int i = 0; i < 32; ++i) {
            int f = i * 128 + tid;                 // 0..4095 float4s
            float4 v = xg[f];
            us4 p = { f2bf(v.x), f2bf(v.y), f2bf(v.z), f2bf(v.w) };
            *(us4*)&xs[f * 4] = p;
        }
    }

    // ---- kick DMA of codebook tile 0 while x staging drains ----
    if (wave == 0) {
        tdm_tile_load(ebf, &es[0][0]);
        __builtin_amdgcn_s_wait_tensorcnt(0);
    }
    __syncthreads();

    // ---- hoist A fragments: this wave's 16 rows, full K=256 ----
    const int rbase = wave * 16;
    const int mrow  = lane & 15;
    const int kbase = (lane >> 4) * 8;      // A 16-bit layout
    const int nb    = lane & 15;            // B column
    const int kbb   = (lane >> 4) * 8;      // B 16-bit layout

    v16bf afr[8];
    #pragma unroll
    for (int t = 0; t < 8; ++t) {
        const int kk = t * 32;
        BFV a;
        a.h[0] = *(const us8*)&xs[(rbase + mrow) * DIM + kk + kbase];
        a.h[1] = *(const us8*)&xs[(rbase + mrow) * DIM + kk + 16 + kbase];
        afr[t] = a.v;
    }

    float bv[8]; int bi[8];
    #pragma unroll
    for (int j = 0; j < 8; ++j) { bv[j] = 3.4e38f; bi[j] = 0; }

    // ---- sweep codebook: DMA tile t+1 overlapped with WMMA on tile t ----
    for (int t = 0; t < NTILES; ++t) {
        if (wave == 0 && t + 1 < NTILES)
            tdm_tile_load(ebf + (size_t)(t + 1) * TILE_ELTS, &es[(t + 1) & 1][0]);

        const unsigned short* eb = &es[t & 1][0];
        const float e0 = esq[t * CT + nb];
        const int   i0 = t * CT + nb;

        // prefetch this lane's full B column for the tile (8 fragments, 256B),
        // so the ds_loads issue back-to-back and drain under the WMMA chain
        BFV b[8];
        #pragma unroll
        for (int k8 = 0; k8 < 8; ++k8) {
            const int kk = k8 * 32;
            b[k8].h[0] = *(const us8*)&eb[nb * DIM + kk + kbb];
            b[k8].h[1] = *(const us8*)&eb[nb * DIM + kk + kbb + 16];
        }

        v8f c = {};
        #pragma unroll
        for (int k8 = 0; k8 < 8; ++k8)
            c = __builtin_amdgcn_wmma_f32_16x16x32_bf16(false, afr[k8], false, b[k8].v,
                                                        (short)0, c, false, false);

        // dist = ||e||^2 - 2*(x.e); keep running (min, first-index) per lane
        #pragma unroll
        for (int j = 0; j < 8; ++j) {
            float d = __builtin_fmaf(-2.0f, c[j], e0);
            if (d < bv[j]) { bv[j] = d; bi[j] = i0; }
        }

        if (wave == 0) __builtin_amdgcn_s_wait_tensorcnt(0);
        __syncthreads();   // tile t fully consumed; tile t+1 resident for all waves
    }

    // ---- cross-lane argmin within 16-lane row-owner groups ----
    #pragma unroll
    for (int j = 0; j < 8; ++j) {
        float v = bv[j]; int idx = bi[j];
        #pragma unroll
        for (int m = 1; m <= 8; m <<= 1) {
            float ov = __shfl_xor(v, m, 32);
            int   oi = __shfl_xor(idx, m, 32);
            if (ov < v || (ov == v && oi < idx)) { v = ov; idx = oi; }
        }
        bv[j] = v; bi[j] = idx;
    }
    if ((lane & 15) == 0) {
        const int rloc = wave * 16 + (lane >> 4) * 8;   // C layout: VGPR j -> M = j + 8*(lane/16)
        #pragma unroll
        for (int j = 0; j < 8; ++j) bidx_s[rloc + j] = bi[j];
    }
    __syncthreads();

    // ---- gather f32 codebook rows, write quantized + indices, f32 loss ----
    float lsum = 0.f;
    for (int rr = 0; rr < 16; ++rr) {
        const int row  = wave * 16 + rr;
        const int code = bidx_s[row];
        const float4* q  = (const float4*)(emb + (size_t)code * DIM);
        const float4* xr = (const float4*)(x + (size_t)(r0 + row) * DIM);
        float4*       o  = (float4*)(out + (size_t)(r0 + row) * DIM);
        #pragma unroll
        for (int c = lane; c < DIM / 4; c += 32) {
            float4 qv = q[c], xv = xr[c];
            o[c] = qv;
            float dx = xv.x - qv.x, dy = xv.y - qv.y;
            float dz = xv.z - qv.z, dw = xv.w - qv.w;
            lsum += dx * dx + dy * dy + dz * dz + dw * dw;
        }
        if (lane == 0) out[(size_t)NQ + 1 + (size_t)(r0 + row)] = (float)code;
    }
    #pragma unroll
    for (int m = 1; m < 32; m <<= 1) lsum += __shfl_xor(lsum, m, 32);
    if (lane == 0) lsum_s[wave] = lsum;
    __syncthreads();
    if (tid == 0)
        partial[blockIdx.x] = lsum_s[0] + lsum_s[1] + lsum_s[2] + lsum_s[3];
}

// ------------------------------------------------------------------ loss ----
__global__ __launch_bounds__(64) void vq_loss(const float* __restrict__ partial,
                                              float* __restrict__ out) {
    if (blockIdx.x == 0 && threadIdx.x == 0) {
        float s = 0.f;
        for (int i = 0; i < NROWS_TOTAL / BROWS; ++i) s += partial[i];  // fixed order
        out[NQ] = 0.25f * s / ((float)NROWS_TOTAL * (float)DIM);
    }
}

// ---------------------------------------------------------------- launch ----
extern "C" void kernel_launch(void* const* d_in, const int* in_sizes, int n_in,
                              void* d_out, int out_size, void* d_ws, size_t ws_size,
                              hipStream_t stream) {
    (void)in_sizes; (void)n_in; (void)out_size; (void)ws_size;
    const float* x   = (const float*)d_in[0];       // [32,2048,256] f32
    const float* emb = (const float*)d_in[1];       // [1024,256]    f32
    float* out = (float*)d_out;                     // quantized | loss | indices(as float)

    unsigned short* ebf = (unsigned short*)d_ws;                    // 512 KB bf16 codebook
    float* esq     = (float*)((char*)d_ws + (size_t)NCODES * DIM * 2);  // 4 KB
    float* partial = esq + NCODES;                                  // 4 KB

    vq_prep<<<NCODES / 16, 128, 0, stream>>>(emb, ebf, esq);
    vq_main<<<NROWS_TOTAL / BROWS, 128, 0, stream>>>(x, emb, ebf, esq, out, partial);
    vq_loss<<<1, 64, 0, stream>>>(partial, out);
}
